// MetaDGCRU_12876311953416
// MI455X (gfx1250) — compile-verified, matrix operand order checked by reading, not run on hardware
//
#include <hip/hip_runtime.h>

// ---------------------------------------------------------------------------
// MetaDGCRU on MI455X (gfx1250): bf16 WMMA, fp32 accumulate, wave32.
//
// Fragment fact (16-bit 16x32 operands): lane (h = lane>>4, c = lane&15) holds
// two contiguous K-runs: K in [8h, 8h+8) and [16+8h, 16+8h+8)  ->  any operand
// stored with K contiguous in memory is loadable as two b128s per lane.
//
// Layouts:
//   G16   bf16 [2][B][400 row][416 K]   row-major graphs (A of hop, direct)
//   HcatT bf16 [B][400 feat][416 node]  K-major features (B of hop, direct)
//   Hnm   bf16 [B][400 node][400 feat]  node-major features (A of meta, direct)
//   Wsw   bf16 [200 kt][NT o][16 col][32 k]  weight B-fragments (async->LDS)
//
// Hop:  no LDS, no barriers; 13 unrolled WMMAs, 4 b128 loads each.
// Meta: no barriers; per-wave async double-buffered weight tiles (ASYNCcnt),
//       A built in registers with incremental d/cc state + v_pk_mul_bf16.
// ---------------------------------------------------------------------------

#define BB     8
#define NND    400
#define NP     416
#define CW     80
#define HROW   400
#define KT_META 200
#define ROWS   (BB*NND)

typedef __attribute__((ext_vector_type(16))) __bf16 bf16x16;
typedef __attribute__((ext_vector_type(8)))  __bf16 bf16x8;
typedef __attribute__((ext_vector_type(8)))  float  floatx8;

union FragB16 { bf16x16 v; unsigned short u[16]; uint4 q[2]; };
union Frag8   { bf16x8  v; unsigned short u[8];  uint4 q;    };
union PackC   { uint4 q; unsigned short u[8]; };

__device__ __forceinline__ unsigned short f2bf(float f) {
    unsigned int u = __float_as_uint(f);
    u += 0x7fffu + ((u >> 16) & 1u);
    return (unsigned short)(u >> 16);
}
__device__ __forceinline__ float bf2f(unsigned short s) {
    return __uint_as_float(((unsigned int)s) << 16);
}

// async copy of one 1KB tile (2 x 16B per lane) global -> LDS (ASYNCcnt).
// INST_OFFSET is added to both LDS and global addresses, so offset:512
// moves both halves of the tile together.
__device__ __forceinline__ void async_copy_tile(const unsigned short* gtile,
                                                unsigned lds_byte, int lane) {
    unsigned voff = (unsigned)(lane * 16);
    unsigned ldsa = lds_byte + voff;
    asm volatile(
        "global_load_async_to_lds_b128 %0, %1, %2\n\t"
        "global_load_async_to_lds_b128 %0, %1, %2 offset:512"
        :: "v"(ldsa), "v"(voff), "s"((unsigned long long)gtile)
        : "memory");
}
__device__ __forceinline__ unsigned lds_addr_of(const void* p) {
    return (unsigned)(uintptr_t)p;      // low 32 bits of generic ptr = LDS offset
}

// ---------------------------------------------------------------------------
// prep: graphs fp32 -> G16 bf16 [2*B][400][416], K zero-padded
// ---------------------------------------------------------------------------
__global__ void prep_g16_kernel(const float* __restrict__ G,
                                unsigned short* __restrict__ G16) {
    const size_t total = (size_t)2 * BB * NND * NP;
    for (size_t e = (size_t)blockIdx.x * blockDim.x + threadIdx.x; e < total;
         e += (size_t)gridDim.x * blockDim.x) {
        int m = (int)(e % NP);
        size_t rest = e / NP;
        int row = (int)(rest % NND);
        int gb  = (int)(rest / NND);
        float v = (m < NND) ? G[((size_t)gb * NND + row) * NND + m] : 0.f;
        G16[e] = f2bf(v);
    }
}

// ---------------------------------------------------------------------------
// prep: both feature layouts = bf16(concat(x,state)) in chunk0, zero elsewhere
// ---------------------------------------------------------------------------
__device__ __forceinline__ float xs_val(const float* x, const float* state,
                                        size_t rowg, int col) {
    if (col < 2)  return x[rowg * 2 + col];
    if (col < 66) return state[rowg * 64 + (col - 2)];
    return 0.f;
}

__global__ void prep_hcat_kernel(const float* __restrict__ x,
                                 const float* __restrict__ state,
                                 unsigned short* __restrict__ HcatT,
                                 unsigned short* __restrict__ Hnm) {
    const size_t stride = (size_t)gridDim.x * blockDim.x;
    const size_t t1 = (size_t)BB * HROW * NP;       // K-major
    for (size_t e = (size_t)blockIdx.x * blockDim.x + threadIdx.x; e < t1; e += stride) {
        int row = (int)(e % NP);
        int col = (int)((e / NP) % HROW);
        int b   = (int)(e / ((size_t)NP * HROW));
        float v = (row < NND && col < 66) ? xs_val(x, state, (size_t)b * NND + row, col) : 0.f;
        HcatT[e] = f2bf(v);
    }
    const size_t t2 = (size_t)BB * NND * HROW;      // node-major
    for (size_t e = (size_t)blockIdx.x * blockDim.x + threadIdx.x; e < t2; e += stride) {
        int col = (int)(e % HROW);
        int n   = (int)((e / HROW) % NND);
        int b   = (int)(e / ((size_t)HROW * NND));
        float v = (col < 66) ? xs_val(x, state, (size_t)b * NND + n, col) : 0.f;
        Hnm[e] = f2bf(v);
    }
}

// ---------------------------------------------------------------------------
// prep: Wsw[kt][o][col(16)][k(32)] = bf16(W[d][i][o*16+col]), K = d*400+cc
// ---------------------------------------------------------------------------
__global__ void prep_wsw_kernel(const float* __restrict__ W,
                                unsigned short* __restrict__ Wsw,
                                int O, int NT) {
    const size_t total = (size_t)KT_META * NT * 512;
    for (size_t e = (size_t)blockIdx.x * blockDim.x + threadIdx.x; e < total;
         e += (size_t)gridDim.x * blockDim.x) {
        int kl  = (int)(e & 31);
        int col = (int)((e >> 5) & 15);
        int o   = (int)((e >> 9) % NT);
        int kt  = (int)(e / ((size_t)512 * NT));
        int k = kt * 32 + kl;
        int d = k / 400, cc = k - d * 400;
        int ch = cc / CW, jc = cc - ch * CW;
        float v = 0.f;
        if (jc < 66)
            v = W[((size_t)d * 330 + (ch * 66 + jc)) * O + (o * 16 + col)];
        Wsw[e] = f2bf(v);
    }
}

// ---------------------------------------------------------------------------
// hop: dst = G @ src.  Pure register kernel: A from G16 rows, B from HcatT
// K-major columns, both as two b128 per lane per K-step. No LDS, no barriers.
// Stores C into BOTH layouts (K-major b128 + node-major scalar).
// ---------------------------------------------------------------------------
__global__ __launch_bounds__(160)
void hop_kernel(const unsigned short* __restrict__ G16,   // this graph: [B][400][416]
                unsigned short* HcatT,
                unsigned short* __restrict__ Hnm,
                int src_chunk, int dst_chunk) {
    const int mt = blockIdx.x, b = blockIdx.y;
    const int wave = threadIdx.x >> 5, lane = threadIdx.x & 31;
    const int half = lane >> 4, l15 = lane & 15;

    const unsigned short* pa0 =
        G16 + ((size_t)b * NND + mt * 16 + l15) * NP + 8 * half;
    const unsigned short* pb0 =
        HcatT + ((size_t)b * HROW + src_chunk * CW + wave * 16 + l15) * NP + 8 * half;

    floatx8 acc = {0.f, 0.f, 0.f, 0.f, 0.f, 0.f, 0.f, 0.f};
#pragma unroll
    for (int kt = 0; kt < 13; ++kt) {
        FragB16 a, bf;
        a.q[0]  = *(const uint4*)(pa0 + kt * 32);
        a.q[1]  = *(const uint4*)(pa0 + kt * 32 + 16);
        bf.q[0] = *(const uint4*)(pb0 + kt * 32);
        bf.q[1] = *(const uint4*)(pb0 + kt * 32 + 16);
        acc = __builtin_amdgcn_wmma_f32_16x16x32_bf16(false, a.v, false, bf.v,
                                                      (short)0, acc, false, false);
    }
    PackC cs;
#pragma unroll
    for (int r = 0; r < 8; ++r) cs.u[r] = f2bf(acc[r]);
    const int dcol = dst_chunk * CW + wave * 16 + l15;
    // K-major: 8 consecutive M at fixed col -> one b128 store
    unsigned short* dT =
        HcatT + ((size_t)b * HROW + dcol) * NP + mt * 16 + 8 * half;
    *(uint4*)dT = cs.q;
    // node-major: scattered u16 stores (once per hop, cheap)
#pragma unroll
    for (int r = 0; r < 8; ++r)
        Hnm[((size_t)b * NND + mt * 16 + 8 * half + r) * HROW + dcol] = cs.u[r];
}

// ---------------------------------------------------------------------------
// meta core: acc = q @ Wsw, q[k] = emb[d]*h[cc] built per-lane in registers.
// Each 8-aligned K-run has a single d (400 % 8 == 0): one splat bf16 multiply.
// d/cc advance incrementally; emb reloaded only on cc wrap. Weight tiles
// stream per-wave: global_load_async_to_lds_b128 + s_wait_asynccnt. No syncs.
// ---------------------------------------------------------------------------
template<int NT>
__device__ __forceinline__ floatx8 meta_acc(const unsigned short* __restrict__ Hnm,
                                            const float* __restrict__ emb,
                                            const unsigned short* __restrict__ Wsw,
                                            unsigned short* sB,     // [2*512] per-wave
                                            int mt) {
    const int tid = threadIdx.x;
    const int o = tid >> 5, lane = tid & 31;
    const int half = lane >> 4, l15 = lane & 15;
    const unsigned sB_byte = lds_addr_of(sB);

    const int grow = mt * 16 + l15;
    const int b = grow / NND, n = grow - b * NND;
    const unsigned short* hrow = Hnm + ((size_t)b * NND + n) * HROW;
    const float* erow = emb + (size_t)grow * 16;

    int cc1 = 8 * half,      d1 = 0;
    int cc2 = 16 + 8 * half, d2 = 0;
    __bf16 ev1 = (__bf16)erow[0];
    __bf16 ev2 = ev1;

    floatx8 acc = {0.f, 0.f, 0.f, 0.f, 0.f, 0.f, 0.f, 0.f};
    async_copy_tile(Wsw + (size_t)o * 512, sB_byte, lane);      // tile 0 -> buf 0
    for (int kt = 0; kt < KT_META; ++kt) {
        const int cur = kt & 1;
        if (kt + 1 < KT_META) {
            async_copy_tile(Wsw + (size_t)((kt + 1) * NT + o) * 512,
                            sB_byte + (unsigned)((cur ^ 1) * 1024), lane);
            asm volatile("s_wait_asynccnt 0x2" ::: "memory");   // tile kt resident
        } else {
            asm volatile("s_wait_asynccnt 0x0" ::: "memory");
        }
        Frag8 r1, r2, p1, p2;
        r1.q = *(const uint4*)(hrow + cc1);
        r2.q = *(const uint4*)(hrow + cc2);
        p1.v = r1.v * ev1;                  // v_pk_mul_bf16 x4 (splat scale)
        p2.v = r2.v * ev2;
        FragB16 a;
        a.q[0] = p1.q;
        a.q[1] = p2.q;
        FragB16 bf;
        const unsigned short* pb = &sB[cur * 512 + l15 * 32 + 8 * half];
        bf.q[0] = *(const uint4*)pb;
        bf.q[1] = *(const uint4*)(pb + 16);
        acc = __builtin_amdgcn_wmma_f32_16x16x32_bf16(false, a.v, false, bf.v,
                                                      (short)0, acc, false, false);
        cc1 += 32;
        if (cc1 >= 400) { cc1 -= 400; ++d1; if (d1 < 16) ev1 = (__bf16)erow[d1]; }
        cc2 += 32;
        if (cc2 >= 400) { cc2 -= 400; ++d2; if (d2 < 16) ev2 = (__bf16)erow[d2]; }
    }
    return acc;
}

__global__ __launch_bounds__(256)
void meta_gate_kernel(const unsigned short* __restrict__ Hnm,
                      const float* __restrict__ emb,
                      const unsigned short* __restrict__ Wsw,
                      const float* __restrict__ bg,      // [16][128]
                      float* __restrict__ ZR) {          // [3200][128]
    __shared__ __align__(16) unsigned short sB[8][2 * 512];
    const int mt = blockIdx.x, tid = threadIdx.x;
    const int o = tid >> 5, lane = tid & 31;
    const int half = lane >> 4, l15 = lane & 15;
    floatx8 acc = meta_acc<8>(Hnm, emb, Wsw, sB[o], mt);
#pragma unroll
    for (int r = 0; r < 8; ++r) {
        int grow = mt * 16 + 8 * half + r;
        int col  = o * 16 + l15;
        float bias = 0.f;
#pragma unroll
        for (int d = 0; d < 16; ++d)
            bias += emb[(size_t)grow * 16 + d] * bg[d * 128 + col];
        float v = acc[r] + bias;
        ZR[(size_t)grow * 128 + col] = 1.f / (1.f + __expf(-v));
    }
}

__global__ __launch_bounds__(128)
void meta_cand_kernel(const unsigned short* __restrict__ Hnm,
                      const float* __restrict__ emb,
                      const unsigned short* __restrict__ Wsw,   // [200][4][512]
                      const float* __restrict__ bc,      // [16][64]
                      const float* __restrict__ ZR,      // [3200][128]
                      const float* __restrict__ state,   // [3200][64]
                      float* __restrict__ out) {         // [3200][64]
    __shared__ __align__(16) unsigned short sB[4][2 * 512];
    const int mt = blockIdx.x, tid = threadIdx.x;
    const int o = tid >> 5, lane = tid & 31;
    const int half = lane >> 4, l15 = lane & 15;
    floatx8 acc = meta_acc<4>(Hnm, emb, Wsw, sB[o], mt);
#pragma unroll
    for (int r = 0; r < 8; ++r) {
        int grow = mt * 16 + 8 * half + r;
        int col  = o * 16 + l15;
        float bias = 0.f;
#pragma unroll
        for (int d = 0; d < 16; ++d)
            bias += emb[(size_t)grow * 16 + d] * bc[d * 64 + col];
        float hc = tanhf(acc[r] + bias);
        float z  = ZR[(size_t)grow * 128 + col];
        float s  = state[(size_t)grow * 64 + col];
        out[(size_t)grow * 64 + col] = z * s + (1.f - z) * hc;
    }
}

// ---------------------------------------------------------------------------
// build xrs into chunk 0 of BOTH layouts: [x | r*state], r = zr[:,64:128]
// ---------------------------------------------------------------------------
__device__ __forceinline__ float xrs_val(const float* x, const float* state,
                                         const float* ZR, size_t rowg, int col) {
    if (col < 2) return x[rowg * 2 + col];
    if (col < 66) {
        int j = col - 2;
        return ZR[rowg * 128 + 64 + j] * state[rowg * 64 + j];
    }
    return 0.f;
}

__global__ void build_xrs_kernel(const float* __restrict__ x,
                                 const float* __restrict__ state,
                                 const float* __restrict__ ZR,
                                 unsigned short* __restrict__ HcatT,
                                 unsigned short* __restrict__ Hnm) {
    const size_t stride = (size_t)gridDim.x * blockDim.x;
    const size_t total = (size_t)BB * CW * NND;
    // K-major (n fast)
    for (size_t e = (size_t)blockIdx.x * blockDim.x + threadIdx.x; e < total; e += stride) {
        int n   = (int)(e % NND);
        int col = (int)((e / NND) % CW);
        int b   = (int)(e / ((size_t)NND * CW));
        float v = xrs_val(x, state, ZR, (size_t)b * NND + n, col);
        HcatT[((size_t)b * HROW + col) * NP + n] = f2bf(v);
    }
    // node-major (col fast)
    for (size_t e = (size_t)blockIdx.x * blockDim.x + threadIdx.x; e < total; e += stride) {
        int col = (int)(e % CW);
        int n   = (int)((e / CW) % NND);
        int b   = (int)(e / ((size_t)NND * CW));
        float v = xrs_val(x, state, ZR, (size_t)b * NND + n, col);
        Hnm[((size_t)b * NND + n) * HROW + col] = f2bf(v);
    }
}

// ---------------------------------------------------------------------------
extern "C" void kernel_launch(void* const* d_in, const int* in_sizes, int n_in,
                              void* d_out, int out_size, void* d_ws, size_t ws_size,
                              hipStream_t stream) {
    const float* x      = (const float*)d_in[0];
    const float* state  = (const float*)d_in[1];
    const float* graphs = (const float*)d_in[2];
    const float* emb    = (const float*)d_in[3];
    const float* Wg     = (const float*)d_in[4];
    const float* bg     = (const float*)d_in[5];
    const float* Wc     = (const float*)d_in[6];
    const float* bc     = (const float*)d_in[7];
    float* out = (float*)d_out;

    char* ws = (char*)d_ws;
    size_t off = 0;
    auto carve = [&](size_t bytes) {
        char* p = ws + off;
        off = (off + bytes + 255) & ~(size_t)255;
        return p;
    };
    unsigned short* G16   = (unsigned short*)carve((size_t)2 * BB * NND * NP * 2); // 5.32 MB
    unsigned short* HcatT = (unsigned short*)carve((size_t)BB * HROW * NP * 2);    //  2.66 MB
    unsigned short* Hnm   = (unsigned short*)carve((size_t)BB * NND * HROW * 2);   //  2.56 MB
    unsigned short* Wg_sw = (unsigned short*)carve((size_t)KT_META * 8 * 512 * 2); //  1.64 MB
    unsigned short* Wc_sw = (unsigned short*)carve((size_t)KT_META * 4 * 512 * 2); //  0.82 MB
    float*          ZR    = (float*)carve((size_t)ROWS * 128 * 4);                 //  1.64 MB
    (void)ws_size; (void)in_sizes; (void)n_in; (void)out_size;

    const unsigned short* G16_0 = G16;
    const unsigned short* G16_1 = G16 + (size_t)BB * NND * NP;

    prep_g16_kernel<<<10400, 256, 0, stream>>>(graphs, G16);
    prep_hcat_kernel<<<5200, 256, 0, stream>>>(x, state, HcatT, Hnm);
    prep_wsw_kernel<<<1600, 256, 0, stream>>>(Wg, Wg_sw, 128, 8);
    prep_wsw_kernel<<<800, 256, 0, stream>>>(Wc, Wc_sw, 64, 4);

    dim3 hgrid(25, BB);
    hop_kernel<<<hgrid, 160, 0, stream>>>(G16_0, HcatT, Hnm, 0, 1);
    hop_kernel<<<hgrid, 160, 0, stream>>>(G16_0, HcatT, Hnm, 1, 2);
    hop_kernel<<<hgrid, 160, 0, stream>>>(G16_1, HcatT, Hnm, 0, 3);
    hop_kernel<<<hgrid, 160, 0, stream>>>(G16_1, HcatT, Hnm, 3, 4);
    meta_gate_kernel<<<KT_META, 256, 0, stream>>>(Hnm, emb, Wg_sw, bg, ZR);

    build_xrs_kernel<<<1000, 256, 0, stream>>>(x, state, ZR, HcatT, Hnm);
    hop_kernel<<<hgrid, 160, 0, stream>>>(G16_0, HcatT, Hnm, 0, 1);
    hop_kernel<<<hgrid, 160, 0, stream>>>(G16_0, HcatT, Hnm, 1, 2);
    hop_kernel<<<hgrid, 160, 0, stream>>>(G16_1, HcatT, Hnm, 0, 3);
    hop_kernel<<<hgrid, 160, 0, stream>>>(G16_1, HcatT, Hnm, 3, 4);
    meta_cand_kernel<<<KT_META, 128, 0, stream>>>(Hnm, emb, Wc_sw, bc, ZR, state, out);
}
